// OptimizedSimEncoderLayer_37383395344611
// MI455X (gfx1250) — matrix operand order repaired and checked by reading, not compiled, not run on hardware
//
#include <hip/hip_runtime.h>
#include <stdint.h>

// ---------------------------------------------------------------------------
// Problem constants
// ---------------------------------------------------------------------------
#define B_    64
#define N_    257
#define D_    1024
#define H_    16
#define HD_   64
#define NP1   256                  // N-1
#define MROWS (B_ * N_)            // 16448
#define ROWS_PAD 16512             // +64 rows slack for padded tile over-reads
#define HID  ((size_t)MROWS * D_)  // elements in one [B,N,D] tensor
#define VTP  288                   // padded n-stride of transposed V / P tiles

// ---------------------------------------------------------------------------
// CDNA5 WMMA types / helpers
// ---------------------------------------------------------------------------
typedef __attribute__((ext_vector_type(16))) __bf16 v16bf;
typedef __attribute__((ext_vector_type(8)))  __bf16 v8bf;
typedef __attribute__((ext_vector_type(8)))  float  v8f;

__device__ __forceinline__ unsigned short f2bf(float f) {
    union { float f; unsigned u; } x; x.f = f;
    return (unsigned short)((x.u + 0x7FFFu + ((x.u >> 16) & 1u)) >> 16);
}
__device__ __forceinline__ v8f zero8() {
    v8f z;
#pragma unroll
    for (int i = 0; i < 8; ++i) z[i] = 0.f;
    return z;
}
__device__ __forceinline__ v8f wmma_bf16(v16bf a, v16bf b, v8f c) {
    // D = A(16x32) * B(32x16) + C  -> v_wmma_f32_16x16x32_bf16
    return __builtin_amdgcn_wmma_f32_16x16x32_bf16(false, a, false, b,
                                                   (short)0, c, false, false);
}

// Async copy of 16 bytes global -> LDS (GLOBAL_LOAD_ASYNC_TO_LDS_B128, ASYNCcnt)
__device__ __forceinline__ void async_copy16(const unsigned short* lds_dst,
                                             const unsigned short* gsrc) {
    unsigned lds_off = (unsigned)(uintptr_t)lds_dst;  // low 32 bits = LDS offset
    unsigned long long ga = (unsigned long long)(uintptr_t)gsrc;
    asm volatile("global_load_async_to_lds_b128 %0, %1, off"
                 :: "v"(lds_off), "v"(ga)
                 : "memory");
}
__device__ __forceinline__ void wait_async0() {
    asm volatile("s_wait_asynccnt 0x0" ::: "memory");
}

// A fragment (16x32 bf16): lane holds row (lane&15); lane-half K split
// {0..15} / {16..31}. Memory is row-major with 'stride' elems.
__device__ __forceinline__ v16bf frag_a(const unsigned short* p, int stride, int lane) {
    int row = lane & 15;
    int k0  = (lane >> 4) * 16;
    const unsigned short* q = p + (size_t)row * stride + k0;
    v8bf lo = *(const v8bf*)(q);
    v8bf hi = *(const v8bf*)(q + 8);
    return __builtin_shufflevector(lo, hi, 0,1,2,3,4,5,6,7,8,9,10,11,12,13,14,15);
}

// B fragment (32x16 bf16): lane holds col (lane&15); lane-half K split
// {0..15} / {16..31}. Memory holds Bt[col][k] (k contiguous), 'stride' elems/col.
__device__ __forceinline__ v16bf frag_b_kc(const unsigned short* p, int stride, int lane) {
    int col = lane & 15;
    int k0  = (lane >> 4) * 16;
    const unsigned short* q = p + (size_t)col * stride + k0;
    v8bf lo = *(const v8bf*)(q);
    v8bf hi = *(const v8bf*)(q + 8);
    return __builtin_shufflevector(lo, hi, 0,1,2,3,4,5,6,7,8,9,10,11,12,13,14,15);
}

__device__ __forceinline__ float block_sum256(float v, float* sm, int t) {
#pragma unroll
    for (int m = 16; m >= 1; m >>= 1) v += __shfl_xor(v, m, 32);
    __syncthreads();
    if ((t & 31) == 0) sm[t >> 5] = v;
    __syncthreads();
    float r = 0.f;
#pragma unroll
    for (int w = 0; w < 8; ++w) r += sm[w];
    return r;
}

// ---------------------------------------------------------------------------
// Conversions / zero fill
// ---------------------------------------------------------------------------
__global__ void cvt_bf16(const float* __restrict__ src,
                         unsigned short* __restrict__ dst, int n) {
    int i = blockIdx.x * 256 + threadIdx.x;
    if (i < n) dst[i] = f2bf(src[i]);
}

// Transposing f32 -> bf16 conversion for weights: dst[n][k] = src[k][n] (D_ x D_)
__global__ __launch_bounds__(256) void cvtT_bf16(const float* __restrict__ src,
                                                 unsigned short* __restrict__ dst) {
    __shared__ float t[32][33];
    int k0 = blockIdx.x * 32, n0 = blockIdx.y * 32;
    int tx = threadIdx.x & 31, ty = threadIdx.x >> 5;  // 32 x 8
#pragma unroll
    for (int i = 0; i < 4; ++i)
        t[ty + i * 8][tx] = src[(size_t)(k0 + ty + i * 8) * D_ + n0 + tx];
    __syncthreads();
#pragma unroll
    for (int i = 0; i < 4; ++i)
        dst[(size_t)(n0 + ty + i * 8) * D_ + k0 + tx] = f2bf(t[tx][ty + i * 8]);
}

__global__ void zero_f32(float* __restrict__ p, size_t n) {
    size_t i = (size_t)blockIdx.x * 1024 + threadIdx.x;
    if (i < n) p[i] = 0.f;
}
__global__ void zero_u32(unsigned* __restrict__ p, size_t n) {
    size_t i = (size_t)blockIdx.x * 1024 + threadIdx.x;
    if (i < n) p[i] = 0u;
}

// ---------------------------------------------------------------------------
// LayerNorm (one block per token) -> bf16
// ---------------------------------------------------------------------------
__global__ __launch_bounds__(256) void ln_bf16(const float* __restrict__ X,
                                               const float* __restrict__ g,
                                               const float* __restrict__ be,
                                               unsigned short* __restrict__ out) {
    __shared__ float sm[8];
    int row = blockIdx.x, t = threadIdx.x;
    const float* x = X + (size_t)row * D_;
    float4 v = *(const float4*)(x + t * 4);
    float s  = v.x + v.y + v.z + v.w;
    float s2 = v.x * v.x + v.y * v.y + v.z * v.z + v.w * v.w;
    s  = block_sum256(s, sm, t);
    s2 = block_sum256(s2, sm, t);
    float mean = s * (1.f / D_);
    float var  = s2 * (1.f / D_) - mean * mean;
    float inv  = rsqrtf(var + 1e-5f);
    unsigned short* o = out + (size_t)row * D_ + t * 4;
    const float* gg = g + t * 4;
    const float* bb = be + t * 4;
    o[0] = f2bf((v.x - mean) * inv * gg[0] + bb[0]);
    o[1] = f2bf((v.y - mean) * inv * gg[1] + bb[1]);
    o[2] = f2bf((v.z - mean) * inv * gg[2] + bb[2]);
    o[3] = f2bf((v.w - mean) * inv * gg[3] + bb[3]);
}

// ---------------------------------------------------------------------------
// Row L2-norm -> bf16 (hn staging for similarity GEMM)
// ---------------------------------------------------------------------------
__global__ __launch_bounds__(256) void hn_bf16(const float* __restrict__ Hd,
                                               unsigned short* __restrict__ out) {
    __shared__ float sm[8];
    int row = blockIdx.x, t = threadIdx.x;
    float4 v = *(const float4*)(Hd + (size_t)row * D_ + t * 4);
    float s2 = v.x * v.x + v.y * v.y + v.z * v.z + v.w * v.w;
    s2 = block_sum256(s2, sm, t);
    float inv = rsqrtf(s2);
    unsigned short* o = out + (size_t)row * D_ + t * 4;
    o[0] = f2bf(v.x * inv); o[1] = f2bf(v.y * inv);
    o[2] = f2bf(v.z * inv); o[3] = f2bf(v.w * inv);
}

// ---------------------------------------------------------------------------
// WMMA GEMM: C[M,Nc] = A[M,K](bf16) @ Wt[Nc,K](bf16, pre-transposed) + bias
// Block tile 64x128, 8 waves, each wave 32x32 (2x2 WMMA frags), BK=32.
// Double-buffered LDS tiles filled with GLOBAL_LOAD_ASYNC_TO_LDS_B128.
// Epilogues: bf16 (*scale), f32 (+residual), or transposed-V bf16 layout.
// ---------------------------------------------------------------------------
__global__ __launch_bounds__(256) void gemm_bf16(
        const unsigned short* __restrict__ A, const unsigned short* __restrict__ Wt,
        const float* __restrict__ bias, const float* __restrict__ resid,
        unsigned short* __restrict__ outb, float* __restrict__ outf,
        unsigned short* __restrict__ vtout,
        int M, int K, int Nc, float scale) {
    __shared__ __align__(16) unsigned short As[2][64 * 40];   // [64][40] row-major
    __shared__ __align__(16) unsigned short Bs[2][128 * 40];  // Bt[n][k], stride 40
    int tid = threadIdx.x, lane = tid & 31, wid = tid >> 5;
    int m0 = blockIdx.x * 64, n0 = blockIdx.y * 128;
    int wm = (wid >> 2) * 32, wn = (wid & 3) * 32;

    v8f acc[2][2];
#pragma unroll
    for (int i = 0; i < 2; ++i)
#pragma unroll
        for (int j = 0; j < 2; ++j) acc[i][j] = zero8();

    // async stage of one (A,B) tile pair into buffer `bufi`
    auto stage = [&](int bufi, int kt) {
        {   // A tile: 64x32, one 16B chunk per thread
            int idx = tid * 8;
            int r = idx >> 5, c = idx & 31;
            async_copy16(&As[bufi][r * 40 + c],
                         A + (size_t)(m0 + r) * K + kt + c);
        }
#pragma unroll
        for (int i = 0; i < 2; ++i) {  // B tile: 128x32 (K-contiguous in Wt)
            int idx = (tid + i * 256) * 8;
            int r = idx >> 5, c = idx & 31;
            async_copy16(&Bs[bufi][r * 40 + c],
                         Wt + (size_t)(n0 + r) * K + kt + c);
        }
    };

    stage(0, 0);
    wait_async0();
    __syncthreads();

    int buf = 0;
    for (int kt = 0; kt < K; kt += 32) {
        if (kt + 32 < K) {
            stage(buf ^ 1, kt + 32);  // overlap next tile with compute
            __builtin_prefetch(A + (size_t)(m0 + (tid >> 2)) * K + kt + 64, 0, 3);
        }
        v16bf a0 = frag_a(&As[buf][(wm + 0) * 40], 40, lane);
        v16bf a1 = frag_a(&As[buf][(wm + 16) * 40], 40, lane);
        v16bf b0 = frag_b_kc(&Bs[buf][(wn + 0) * 40], 40, lane);
        v16bf b1 = frag_b_kc(&Bs[buf][(wn + 16) * 40], 40, lane);
        acc[0][0] = wmma_bf16(a0, b0, acc[0][0]);
        acc[0][1] = wmma_bf16(a0, b1, acc[0][1]);
        acc[1][0] = wmma_bf16(a1, b0, acc[1][0]);
        acc[1][1] = wmma_bf16(a1, b1, acc[1][1]);
        wait_async0();
        __syncthreads();
        buf ^= 1;
    }

    int col = lane & 15, hi = lane >> 4;
#pragma unroll
    for (int mi = 0; mi < 2; ++mi)
#pragma unroll
        for (int ni = 0; ni < 2; ++ni)
#pragma unroll
            for (int r = 0; r < 8; ++r) {
                int m = m0 + wm + mi * 16 + r + hi * 8;
                int n = n0 + wn + ni * 16 + col;
                float v = acc[mi][ni][r] + bias[n];
                if (outf) {
                    outf[(size_t)m * Nc + n] = v + resid[(size_t)m * Nc + n];
                } else if (vtout) {
                    // transposed V layout: Vt[(b*H+h)*HD+d][n], stride VTP
                    int bb = m / N_, nrow = m - bb * N_;
                    int hh = n >> 6, dd = n & 63;
                    vtout[((size_t)(bb * H_ + hh) * HD_ + dd) * VTP + nrow] = f2bf(v);
                } else {
                    outb[(size_t)m * Nc + n] = f2bf(v * scale);
                }
            }
}

// ---------------------------------------------------------------------------
// Fused attention: one block per (b,h); 4 waves, each owns q-tiles.
// Scores via WMMA (QK^T), softmax in registers with half-wave shuffles,
// normalized P staged in LDS, then P@Vt via WMMA. CLS row -> cls_attn.
// ---------------------------------------------------------------------------
__global__ __launch_bounds__(128) void attn_fused(
        const unsigned short* __restrict__ Q, const unsigned short* __restrict__ Kb,
        const unsigned short* __restrict__ Vt, unsigned short* __restrict__ O,
        float* __restrict__ cls_attn) {
    int b = blockIdx.x >> 4;
    int h = blockIdx.x & 15;
    int lane = threadIdx.x & 31, wid = threadIdx.x >> 5;
    int col = lane & 15, hi = lane >> 4;
    __shared__ __align__(16) unsigned short Ps[4][16 * VTP];  // per-wave P tile

    const size_t rowbase = (size_t)b * N_ * D_;
    const unsigned short* qb  = Q  + rowbase + h * HD_;
    const unsigned short* kb  = Kb + rowbase + h * HD_;
    const unsigned short* vtb = Vt + (size_t)(b * H_ + h) * HD_ * VTP;

    for (int qt = wid; qt < 17; qt += 4) {
        v16bf qa0 = frag_a(qb + (size_t)qt * 16 * D_, D_, lane);
        v16bf qa1 = frag_a(qb + (size_t)qt * 16 * D_ + 32, D_, lane);

        v8f s[17];
#pragma unroll
        for (int nt = 0; nt < 17; ++nt) {
            v8f a = zero8();
            v16bf k0f = frag_b_kc(kb + (size_t)nt * 16 * D_, D_, lane);
            v16bf k1f = frag_b_kc(kb + (size_t)nt * 16 * D_ + 32, D_, lane);
            a = wmma_bf16(qa0, k0f, a);
            a = wmma_bf16(qa1, k1f, a);
            s[nt] = a;
        }
        // mask n >= 257 (tile 16, cols >= 1)
        if (col >= 1) {
#pragma unroll
            for (int r = 0; r < 8; ++r) s[16][r] = -1e30f;
        }
        // per-row softmax; C-layout rows live across 16 lanes of each half
#pragma unroll
        for (int r = 0; r < 8; ++r) {
            float mx = -1e30f;
#pragma unroll
            for (int nt = 0; nt < 17; ++nt) mx = fmaxf(mx, s[nt][r]);
#pragma unroll
            for (int msk = 1; msk < 16; msk <<= 1) mx = fmaxf(mx, __shfl_xor(mx, msk, 32));
            float sum = 0.f;
#pragma unroll
            for (int nt = 0; nt < 17; ++nt) {
                float e = __expf(s[nt][r] - mx);
                s[nt][r] = e;
                sum += e;
            }
#pragma unroll
            for (int msk = 1; msk < 16; msk <<= 1) sum += __shfl_xor(sum, msk, 32);
            float inv = 1.f / sum;
#pragma unroll
            for (int nt = 0; nt < 17; ++nt) {
                float p = s[nt][r] * inv;
                s[nt][r] = p;
                Ps[wid][(r + hi * 8) * VTP + nt * 16 + col] = f2bf(p);
            }
        }
        // zero pad cols 272..287 (K-pad of second WMMA)
#pragma unroll
        for (int r = 0; r < 8; ++r)
            Ps[wid][(r + hi * 8) * VTP + 272 + col] = 0;

        // CLS attention mass (row q=0), summed across heads via atomics
        if (qt == 0 && hi == 0) {
#pragma unroll
            for (int nt = 0; nt < 17; ++nt) {
                int n = nt * 16 + col;
                if (n >= 1 && n < N_)
                    atomicAdd(cls_attn + b * NP1 + (n - 1), s[nt][0]);
            }
        }

        // O(16x64) = P(16x288) @ V(288x64)  (Vt is n-contiguous per d row)
#pragma unroll
        for (int dt = 0; dt < 4; ++dt) {
            v8f oacc = zero8();
#pragma unroll
            for (int kc = 0; kc < 9; ++kc) {
                v16bf pa = frag_a(Ps[wid] + kc * 32, VTP, lane);
                v16bf vf = frag_b_kc(vtb + (size_t)(dt * 16) * VTP + kc * 32, VTP, lane);
                oacc = wmma_bf16(pa, vf, oacc);
            }
#pragma unroll
            for (int r = 0; r < 8; ++r) {
                int qrow = qt * 16 + r + hi * 8;
                if (qrow < N_)
                    O[rowbase + (size_t)qrow * D_ + h * HD_ + dt * 16 + col] =
                        f2bf(oacc[r]);
            }
        }
    }
}

// ---------------------------------------------------------------------------
// Similarity: per block (b, 16-row tile of patches): WMMA over ref bank,
// running max/argmax in registers, half-wave shuffle argmax reduction.
// ---------------------------------------------------------------------------
__global__ __launch_bounds__(32) void sim_kernel(
        const unsigned short* __restrict__ HN, const unsigned short* __restrict__ REF,
        float* __restrict__ best, int* __restrict__ besti) {
    int b = blockIdx.x >> 4, qt = blockIdx.x & 15;
    int lane = threadIdx.x;
    int col = lane & 15, hi = lane >> 4;
    const unsigned short* abase = HN + ((size_t)b * N_ + 1 + qt * 16) * D_;
    const unsigned short* rbase = REF + (size_t)b * N_ * D_;

    float bv[8]; int bi[8];
#pragma unroll
    for (int r = 0; r < 8; ++r) { bv[r] = -1e30f; bi[r] = 0; }

    for (int nt = 0; nt < 17; ++nt) {
        v8f acc = zero8();
#pragma unroll
        for (int kc = 0; kc < 32; ++kc) {
            v16bf a  = frag_a(abase + kc * 32, D_, lane);
            v16bf bb = frag_b_kc(rbase + (size_t)nt * 16 * D_ + kc * 32, D_, lane);
            acc = wmma_bf16(a, bb, acc);
        }
        int n = nt * 16 + col;
        if (n < N_) {
#pragma unroll
            for (int r = 0; r < 8; ++r) {
                if (acc[r] > bv[r]) { bv[r] = acc[r]; bi[r] = n; }
            }
        }
    }
    // argmax across 16 lanes of each half (first-max tie break)
#pragma unroll
    for (int msk = 1; msk < 16; msk <<= 1) {
#pragma unroll
        for (int r = 0; r < 8; ++r) {
            float ov = __shfl_xor(bv[r], msk, 32);
            int   oi = __shfl_xor(bi[r], msk, 32);
            if (ov > bv[r] || (ov == bv[r] && oi < bi[r])) { bv[r] = ov; bi[r] = oi; }
        }
    }
    if (col == 0) {
#pragma unroll
        for (int r = 0; r < 8; ++r) {
            int p = qt * 16 + r + hi * 8;       // patch index 0..255
            best[b * NP1 + p]  = bv[r];
            besti[b * NP1 + p] = bi[r] + b * N_;
        }
    }
}

// ---------------------------------------------------------------------------
// Rank by CLS attention, threshold lookup, reuse map (stable-argsort ranks)
// ---------------------------------------------------------------------------
__global__ __launch_bounds__(256) void rank_kernel(
        const float* __restrict__ cls_attn, const float* __restrict__ thrs,
        const float* __restrict__ best, float* __restrict__ reuse_out,
        int* __restrict__ comp) {
    __shared__ float a[256];
    int b = blockIdx.x, i = threadIdx.x;
    a[i] = cls_attn[b * NP1 + i];
    __syncthreads();
    float ai = a[i];
    int rank = 0;
#pragma unroll 8
    for (int j = 0; j < 256; ++j)
        rank += (a[j] < ai) || (a[j] == ai && j < i);
    float thr = thrs[rank];
    bool reuse = best[b * NP1 + i] >= thr;
    reuse_out[b * NP1 + i] = reuse ? 1.f : 0.f;
    comp[b * NP1 + i] = reuse ? 0 : 1;
}

// ---------------------------------------------------------------------------
// Inclusive scan of 16384 compute flags -> pos = cumsum-1 (single block)
// ---------------------------------------------------------------------------
__global__ __launch_bounds__(1024) void scan_kernel(const int* __restrict__ flags,
                                                    int* __restrict__ pos) {
    __shared__ int arr[1024];
    int t = threadIdx.x;
    int base = t * 16;
    int loc[16];
    int s = 0;
#pragma unroll
    for (int k = 0; k < 16; ++k) { s += flags[base + k]; loc[k] = s; }
    arr[t] = s;
    __syncthreads();
    for (int ofs = 1; ofs < 1024; ofs <<= 1) {
        int add = (t >= ofs) ? arr[t - ofs] : 0;
        __syncthreads();
        arr[t] += add;
        __syncthreads();
    }
    int excl = arr[t] - s;
#pragma unroll
    for (int k = 0; k < 16; ++k) pos[base + k] = excl + loc[k] - 1;
}

// ---------------------------------------------------------------------------
// Scatter computed tokens (+ CLS rows) into compute_states[_norm]
// ---------------------------------------------------------------------------
__global__ __launch_bounds__(256) void scatter_rows(
        const float* __restrict__ Hd, const int* __restrict__ comp,
        const int* __restrict__ pos, float* __restrict__ cs,
        float* __restrict__ csn) {
    __shared__ float sm[8];
    int row = blockIdx.x, t = threadIdx.x;
    int b = row / N_, n = row % N_;
    int dest;
    if (n == 0) dest = b;
    else {
        int j = b * NP1 + n - 1;
        dest = comp[j] ? (B_ + pos[j]) : -1;
    }
    float4 v = *(const float4*)(Hd + (size_t)row * D_ + t * 4);
    float s2 = v.x * v.x + v.y * v.y + v.z * v.z + v.w * v.w;
    s2 = block_sum256(s2, sm, t);
    if (dest < 0) return;
    float inv = rsqrtf(s2);
    float* o1 = cs  + (size_t)dest * D_ + t * 4;
    float* o2 = csn + (size_t)dest * D_ + t * 4;
    o1[0] = v.x; o1[1] = v.y; o1[2] = v.z; o1[3] = v.w;
    o2[0] = v.x * inv; o2[1] = v.y * inv; o2[2] = v.z * inv; o2[3] = v.w * inv;
}

// ---------------------------------------------------------------------------
// Gather-index construction
// ---------------------------------------------------------------------------
__global__ void gather_kernel(const int* __restrict__ pos,
                              const int* __restrict__ comp,
                              const int* __restrict__ besti,
                              int* __restrict__ gout) {
    int i = blockIdx.x * 256 + threadIdx.x;
    if (i >= MROWS) return;
    int b = i / N_, n = i % N_;
    int gi;
    if (n == 0) gi = B_ * N_ + b;
    else {
        int j = b * NP1 + n - 1;
        gi = comp[j] ? (B_ * N_ + B_ + pos[j]) : besti[j];
    }
    gout[i] = gi;
}

// ---------------------------------------------------------------------------
// Host launcher
// ---------------------------------------------------------------------------
extern "C" void kernel_launch(void* const* d_in, const int* in_sizes, int n_in,
                              void* d_out, int out_size, void* d_ws, size_t ws_size,
                              hipStream_t stream) {
    const float* hs    = (const float*)d_in[0];
    const float* refn  = (const float*)d_in[1];
    const float* g1    = (const float*)d_in[2];
    const float* b1    = (const float*)d_in[3];
    const float* wq    = (const float*)d_in[4];
    const float* bq    = (const float*)d_in[5];
    const float* wk    = (const float*)d_in[6];
    const float* bk    = (const float*)d_in[7];
    const float* wv    = (const float*)d_in[8];
    const float* bv    = (const float*)d_in[9];
    const float* wo    = (const float*)d_in[10];
    const float* bo    = (const float*)d_in[11];
    const float* thrs  = (const float*)d_in[12];

    // workspace layout (big bf16 buffers aliased across phases)
    uint8_t* w = (uint8_t*)d_ws;
    const size_t BIGSZ = (size_t)40 * 1024 * 1024;
    unsigned short* xln  = (unsigned short*)(w);             // later: o buffer
    unsigned short* qbuf = (unsigned short*)(w + BIGSZ);     // later: hn bf16
    unsigned short* kbuf = (unsigned short*)(w + 2 * BIGSZ); // later: ref bf16
    unsigned short* vt   = (unsigned short*)(w + 3 * BIGSZ); // transposed V
    unsigned short* wqb  = (unsigned short*)(w + 4 * BIGSZ);
    unsigned short* wkb  = wqb + (size_t)D_ * D_;
    unsigned short* wvb  = wkb + (size_t)D_ * D_;
    unsigned short* wob  = wvb + (size_t)D_ * D_;
    float* cls   = (float*)(wob + (size_t)D_ * D_);
    float* best  = cls + B_ * NP1;
    int*   besti = (int*)(best + B_ * NP1);
    int*   comp  = besti + B_ * NP1;
    int*   pos   = comp + B_ * NP1;

    float* out    = (float*)d_out;
    float* hidden = out;                       // [B,N,D]
    float* cs     = out + HID;                 // compute_states
    float* csn    = out + 2 * HID;             // compute_states_norm
    float* reuse  = out + 3 * HID;             // reuse_map (0/1)
    int*   gout   = (int*)(out + 3 * HID + B_ * NP1);  // gather_idx

    // 1) weights -> bf16, pre-transposed to Wt[n][k] (K-contiguous B tiles)
    dim3 tg(D_ / 32, D_ / 32);
    cvtT_bf16<<<tg, 256, 0, stream>>>(wq, wqb);
    cvtT_bf16<<<tg, 256, 0, stream>>>(wk, wkb);
    cvtT_bf16<<<tg, 256, 0, stream>>>(wv, wvb);
    cvtT_bf16<<<tg, 256, 0, stream>>>(wo, wob);

    // 2) layernorm -> bf16
    ln_bf16<<<MROWS, 256, 0, stream>>>(hs, g1, b1, xln);

    // 3) QKV projections (WMMA GEMM); Q pre-scaled by HD^-0.5.
    //    V is written directly in transposed Vt[(b,h,d)][n] layout (zeroed
    //    first so n-padding contributes exact zeros in P@V).
    const size_t vt_elems = (size_t)B_ * H_ * HD_ * VTP;
    zero_u32<<<(unsigned)((vt_elems / 2 + 1023) / 1024), 1024, 0, stream>>>(
        (unsigned*)vt, vt_elems / 2);
    dim3 gg(MROWS / 64, D_ / 128);
    gemm_bf16<<<gg, 256, 0, stream>>>(xln, wqb, bq, nullptr, qbuf, nullptr, nullptr,
                                      MROWS, D_, D_, 0.125f);
    gemm_bf16<<<gg, 256, 0, stream>>>(xln, wkb, bk, nullptr, kbuf, nullptr, nullptr,
                                      MROWS, D_, D_, 1.0f);
    gemm_bf16<<<gg, 256, 0, stream>>>(xln, wvb, bv, nullptr, nullptr, nullptr, vt,
                                      MROWS, D_, D_, 1.0f);

    // 4) attention (xln region reused as O buffer)
    unsigned short* obuf = xln;
    zero_f32<<<(B_ * NP1) / 1024, 1024, 0, stream>>>(cls, (size_t)B_ * NP1);
    attn_fused<<<B_ * H_, 128, 0, stream>>>(qbuf, kbuf, vt, obuf, cls);

    // 5) output projection + residual -> hidden (f32, into d_out)
    gemm_bf16<<<gg, 256, 0, stream>>>(obuf, wob, bo, hs, nullptr, hidden, nullptr,
                                      MROWS, D_, D_, 1.0f);

    // 6) hn (bf16) for similarity; ref -> bf16 (reuse q/k regions)
    unsigned short* hnb  = qbuf;
    unsigned short* refb = kbuf;
    hn_bf16<<<MROWS, 256, 0, stream>>>(hidden, hnb);
    cvt_bf16<<<(int)(HID / 256), 256, 0, stream>>>(refn, refb, (int)HID);

    // 7) similarity search (WMMA + argmax)
    sim_kernel<<<B_ * 16, 32, 0, stream>>>(hnb, refb, best, besti);

    // 8) ranks/thresholds -> reuse map + compute flags
    rank_kernel<<<B_, 256, 0, stream>>>(cls, thrs, best, reuse, comp);

    // 9) scan -> compacted positions
    scan_kernel<<<1, 1024, 0, stream>>>(comp, pos);

    // 10) zero + scatter compute_states / compute_states_norm
    zero_f32<<<(unsigned)((2 * HID + 1023) / 1024), 1024, 0, stream>>>(cs, 2 * HID);
    scatter_rows<<<MROWS, 256, 0, stream>>>(hidden, comp, pos, cs, csn);

    // 11) gather indices
    gather_kernel<<<(MROWS + 255) / 256, 256, 0, stream>>>(pos, comp, besti, gout);
}
